// MoELLMModel_2697239462426
// MI455X (gfx1250) — matrix operand order repaired and checked by reading, not compiled, not run on hardware
//
#include <hip/hip_runtime.h>

// MoE forward for MI455X (gfx1250, wave32, WMMA bf16 path).
// T=8192 tokens, H=1024, F=4096, E=8, top-2 routing.
// Round 3: async Global->LDS (GLOBAL_LOAD_ASYNC_TO_LDS_B128) for the bf16
// down-proj A tile + s_wait_asynccnt fencing; packed bf16 cvt when available.

#define T_TOK 8192
#define H_DIM 1024
#define F_DIM 4096
#define E_NUM 8

#define BM 128
#define BN1 64     // gate/up N-tile
#define BN2 128    // down N-tile
#define KC 32

typedef __bf16 bf16;
typedef __attribute__((ext_vector_type(16))) __bf16 v16bf;
typedef __attribute__((ext_vector_type(2)))  __bf16 v2bf;
typedef __attribute__((ext_vector_type(8)))  float  v8f;

__device__ __forceinline__ unsigned pack2(float a, float b) {
#if __has_builtin(__builtin_amdgcn_cvt_pk_bf16_f32)
  union { v2bf v; unsigned u; } r;
  r.v = __builtin_amdgcn_cvt_pk_bf16_f32(a, b);
  return r.u;
#else
  union { bf16 h[2]; unsigned u; } r;
  r.h[0] = (bf16)a; r.h[1] = (bf16)b;
  return r.u;
#endif
}

// Generic -> LDS(addrspace 3) byte address for async-copy destinations.
__device__ __forceinline__ unsigned lds_addr(const void* p) {
  return (unsigned)(unsigned long long)(const __attribute__((address_space(3))) void*)p;
}

// ---------------------------------------------------------------------------
// Router: one wave per token. logits = x @ w_gate ([T,8]); softmax; top-2;
// renormalized combine weights cw[T,8] (0 for unrouted experts).
// ---------------------------------------------------------------------------
__global__ __launch_bounds__(256) void router_kernel(
    const float* __restrict__ x,        // [T,H]
    const float* __restrict__ wg,       // [H,E]
    float* __restrict__ cw,             // [T,E] (workspace)
    float* __restrict__ logits_out)     // [T,E] (tail of d_out)
{
  const int wave = threadIdx.x >> 5;
  const int lane = threadIdx.x & 31;
  const int t = blockIdx.x * 8 + wave;

  float acc[E_NUM];
#pragma unroll
  for (int e = 0; e < E_NUM; ++e) acc[e] = 0.f;

  const float* xr = x + (size_t)t * H_DIM;
  for (int i = lane; i < H_DIM; i += 32) {
    float xv = xr[i];
    const float* wr = wg + (size_t)i * E_NUM;
#pragma unroll
    for (int e = 0; e < E_NUM; ++e) acc[e] += xv * wr[e];
  }
#pragma unroll
  for (int off = 16; off > 0; off >>= 1) {
#pragma unroll
    for (int e = 0; e < E_NUM; ++e)
      acc[e] += __shfl_xor(acc[e], off, 32);
  }

  if (lane == 0) {
    float m = acc[0];
#pragma unroll
    for (int e = 1; e < E_NUM; ++e) m = fmaxf(m, acc[e]);
    float p[E_NUM];
#pragma unroll
    for (int e = 0; e < E_NUM; ++e) p[e] = __expf(acc[e] - m);
    int i0 = 0;
#pragma unroll
    for (int e = 1; e < E_NUM; ++e) if (p[e] > p[i0]) i0 = e;
    int i1 = (i0 == 0) ? 1 : 0;
#pragma unroll
    for (int e = 0; e < E_NUM; ++e) if (e != i0 && p[e] > p[i1]) i1 = e;
    float denom = __builtin_amdgcn_rcpf(p[i0] + p[i1]);

    float* cwr = cw + (size_t)t * E_NUM;
#pragma unroll
    for (int e = 0; e < E_NUM; ++e) cwr[e] = 0.f;
    cwr[i0] = p[i0] * denom;
    cwr[i1] = p[i1] * denom;

    float* lr = logits_out + (size_t)t * E_NUM;
#pragma unroll
    for (int e = 0; e < E_NUM; ++e) lr[e] = acc[e];
  }
}

// ---------------------------------------------------------------------------
// Fragment-swizzled LDS layouts (wave32 WMMA bf16, 16x16x32):
//  A (16x32):  lane = (m%16) + 16*((k%16)/8) ; elem j = (k/16)*8 + (k%8)
//  B (32x16):  lane = n + 16*((k%16)/8)       ; elem j = (k/16)*8 + (k%8)
//  C/D (f32):  M = v + 8*(lane/16), N = lane%16
// Per-lane 16-bf16 fragments are contiguous (32 B) -> 2x ds_load_b128 each.
// K-adjacent groups stay contiguous within a lane -> 16B-granular scatter,
// which matches GLOBAL_LOAD_ASYNC_TO_LDS_B128 (per-lane LDS dest address).
// ---------------------------------------------------------------------------

// gate/up GEMM for one expert: act[t,f] = silu(x@Wg) * (x@Wu), bf16 out.
__global__ __launch_bounds__(256) void gate_up_kernel(
    const float* __restrict__ x,        // [T,H]
    const float* __restrict__ wgp,      // [H,F] (expert slice)
    const float* __restrict__ wup,      // [H,F]
    bf16* __restrict__ act)             // [T,F]
{
  __shared__ alignas(32) bf16 aLds[2][BM / 16][32][16];   // 16 KB
  __shared__ alignas(32) bf16 bLdsG[2][BN1 / 16][32][16]; // 8 KB
  __shared__ alignas(32) bf16 bLdsU[2][BN1 / 16][32][16]; // 8 KB

  const int tid  = threadIdx.x;
  const int lane = tid & 31;
  const int wave = tid >> 5;
  const int t0 = blockIdx.x * BM;
  const int f0 = blockIdx.y * BN1;

  const int mtBase = (wave & 3) * 2;   // 2 of 8 m-tiles
  const int ntBase = (wave >> 2) * 2;  // 2 of 4 n-tiles

  v8f accG[2][2], accU[2][2];
#pragma unroll
  for (int i = 0; i < 2; ++i)
#pragma unroll
    for (int j = 0; j < 2; ++j)
#pragma unroll
      for (int v = 0; v < 8; ++v) { accG[i][j][v] = 0.f; accU[i][j][v] = 0.f; }

  auto load_tile = [&](int buf, int k0) {
    // A: 128x32 f32 -> bf16; 8-wide k-chunks: 2x float4 load, 1x b128 store.
#pragma unroll
    for (int i = 0; i < 2; ++i) {
      int idx = tid + i * 256;               // 0..511
      int r  = idx >> 2;                     // token row 0..127
      int cg = (idx & 3) * 8;                // k-chunk base 0,8,16,24
      const float* src = &x[(size_t)(t0 + r) * H_DIM + (k0 + cg)];
      float4 v0 = *(const float4*)(src);
      float4 v1 = *(const float4*)(src + 4);
      int mt = r >> 4, lm = r & 15;
      int ln = lm + (((cg >> 3) & 1) << 4);
      int j  = (cg >> 4) << 3;               // 0 or 8
      uint4 pk;
      pk.x = pack2(v0.x, v0.y); pk.y = pack2(v0.z, v0.w);
      pk.z = pack2(v1.x, v1.y); pk.w = pack2(v1.z, v1.w);
      *(uint4*)(&aLds[buf][mt][ln][j]) = pk;
    }
    // B (gate & up): 32x64 each; k-pairs: 2x b32 load, 1x b32 store.
#pragma unroll
    for (int i = 0; i < 4; ++i) {
      int idx = tid + i * 256;               // 0..1023
      int n  = idx & 63;
      int rp = (idx >> 6) * 2;               // even k row 0..30
      size_t g0 = (size_t)(k0 + rp) * F_DIM + (f0 + n);
      float ga = wgp[g0], gb = wgp[g0 + F_DIM];
      float ua = wup[g0], ub = wup[g0 + F_DIM];
      int nt = n >> 4, nl = n & 15;
      int ln = nl + (((rp >> 3) & 1) << 4);
      int j  = ((rp >> 4) << 3) + (rp & 7);  // even
      *(unsigned*)(&bLdsG[buf][nt][ln][j]) = pack2(ga, gb);
      *(unsigned*)(&bLdsU[buf][nt][ln][j]) = pack2(ua, ub);
    }
  };

  auto compute = [&](int buf) {
    v16bf aF[2], bG[2], bU[2];
#pragma unroll
    for (int i = 0; i < 2; ++i)
      aF[i] = *(const v16bf*)(&aLds[buf][mtBase + i][lane][0]);
#pragma unroll
    for (int j = 0; j < 2; ++j) {
      bG[j] = *(const v16bf*)(&bLdsG[buf][ntBase + j][lane][0]);
      bU[j] = *(const v16bf*)(&bLdsU[buf][ntBase + j][lane][0]);
    }
#pragma unroll
    for (int i = 0; i < 2; ++i)
#pragma unroll
      for (int j = 0; j < 2; ++j) {
        accG[i][j] = __builtin_amdgcn_wmma_f32_16x16x32_bf16(
            false, aF[i], false, bG[j], (short)0, accG[i][j], false, false);
        accU[i][j] = __builtin_amdgcn_wmma_f32_16x16x32_bf16(
            false, aF[i], false, bU[j], (short)0, accU[i][j], false, false);
      }
  };

  load_tile(0, 0);
  int buf = 0;
  const int NK = H_DIM / KC;
  for (int kt = 0; kt < NK; ++kt) {
    __syncthreads();
    if (kt + 1 < NK) load_tile(buf ^ 1, (kt + 1) * KC);
    compute(buf);
    buf ^= 1;
  }

  // epilogue: act = silu(g) * u   (silu via v_rcp_f32)
  const int hf = lane >> 4;
  const int nl = lane & 15;
#pragma unroll
  for (int i = 0; i < 2; ++i)
#pragma unroll
    for (int j = 0; j < 2; ++j)
#pragma unroll
      for (int v = 0; v < 8; ++v) {
        int t = t0 + (mtBase + i) * 16 + v + hf * 8;
        int f = f0 + (ntBase + j) * 16 + nl;
        float g = accG[i][j][v];
        float u = accU[i][j][v];
        float a = g * __builtin_amdgcn_rcpf(1.f + __expf(-g)) * u;
        act[(size_t)t * F_DIM + f] = (bf16)a;
      }
}

// down GEMM for one expert: out[t,h] (+)= cw[t,e] * (act @ Wd)[t,h]
__global__ __launch_bounds__(256) void down_kernel(
    const bf16* __restrict__ act,       // [T,F] bf16
    const float* __restrict__ wd,       // [F,H] (expert slice)
    const float* __restrict__ cw,       // [T,E]
    int expert, int first,
    float* __restrict__ out)            // [T,H]
{
  __shared__ alignas(32) bf16 aLds[2][BM / 16][32][16];   // 16 KB
  __shared__ alignas(32) bf16 bLds[2][BN2 / 16][32][16];  // 16 KB
  __shared__ float cwLds[BM];

  const int tid  = threadIdx.x;
  const int lane = tid & 31;
  const int wave = tid >> 5;
  const int t0 = blockIdx.x * BM;
  const int h0 = blockIdx.y * BN2;

  const int mtBase = (wave & 3) * 2;   // 2 of 8 m-tiles
  const int ntBase = (wave >> 2) * 4;  // 4 of 8 n-tiles

  if (tid < BM) cwLds[tid] = cw[(size_t)(t0 + tid) * E_NUM + expert];

  v8f acc[2][4];
#pragma unroll
  for (int i = 0; i < 2; ++i)
#pragma unroll
    for (int j = 0; j < 4; ++j)
#pragma unroll
      for (int v = 0; v < 8; ++v) acc[i][j][v] = 0.f;

  // Precompute per-thread async-copy descriptors for the A tile:
  // 2 chunks x 16 B, fragment-swizzled scatter into LDS, source is bf16 act.
  unsigned ldsA[2][2];           // [chunk][buf]
  const bf16* gA[2];
#pragma unroll
  for (int i = 0; i < 2; ++i) {
    int idx = tid + i * 256;     // 0..511
    int r  = idx >> 2;
    int cg = (idx & 3) * 8;
    int mt = r >> 4, lm = r & 15;
    int ln = lm + (((cg >> 3) & 1) << 4);
    int j  = (cg >> 4) << 3;
    ldsA[i][0] = lds_addr(&aLds[0][mt][ln][j]);
    ldsA[i][1] = lds_addr(&aLds[1][mt][ln][j]);
    gA[i] = &act[(size_t)(t0 + r) * F_DIM + cg];
  }

  auto issueA = [&](int buf, int k0) {
    // Async DMA: per-lane 16B global -> LDS (fragment position), no VGPR
    // round-trip; tracked on ASYNCcnt.
#pragma unroll
    for (int i = 0; i < 2; ++i) {
      unsigned long long src = (unsigned long long)(size_t)(gA[i] + k0);
      asm volatile("global_load_async_to_lds_b128 %0, %1, off"
                   :: "v"(ldsA[i][buf]), "v"(src)
                   : "memory");
    }
  };

  auto loadB = [&](int buf, int k0) {
    // B: 32x128 f32 -> bf16; k-pairs.
#pragma unroll
    for (int i = 0; i < 8; ++i) {
      int idx = tid + i * 256;               // 0..2047
      int n  = idx & 127;
      int rp = (idx >> 7) * 2;               // even k row 0..30
      size_t g0 = (size_t)(k0 + rp) * H_DIM + (h0 + n);
      float a = wd[g0], b = wd[g0 + H_DIM];
      int nt = n >> 4, nl = n & 15;
      int ln = nl + (((rp >> 3) & 1) << 4);
      int j  = ((rp >> 4) << 3) + (rp & 7);
      *(unsigned*)(&bLds[buf][nt][ln][j]) = pack2(a, b);
    }
  };

  auto compute = [&](int buf) {
    v16bf aF[2], bF[4];
#pragma unroll
    for (int i = 0; i < 2; ++i)
      aF[i] = *(const v16bf*)(&aLds[buf][mtBase + i][lane][0]);
#pragma unroll
    for (int j = 0; j < 4; ++j)
      bF[j] = *(const v16bf*)(&bLds[buf][ntBase + j][lane][0]);
#pragma unroll
    for (int i = 0; i < 2; ++i)
#pragma unroll
      for (int j = 0; j < 4; ++j)
        acc[i][j] = __builtin_amdgcn_wmma_f32_16x16x32_bf16(
            false, aF[i], false, bF[j], (short)0, acc[i][j], false, false);
  };

  issueA(0, 0);
  loadB(0, 0);
  int buf = 0;
  const int NK = F_DIM / KC;
  for (int kt = 0; kt < NK; ++kt) {
    // All of this wave's async LDS writes (for `buf`) must land before the
    // publishing barrier; ds stores are waited by the barrier lowering itself.
    asm volatile("s_wait_asynccnt 0" ::: "memory");
    __syncthreads();
    if (kt + 1 < NK) {
      issueA(buf ^ 1, (kt + 1) * KC);
      loadB(buf ^ 1, (kt + 1) * KC);
    }
    compute(buf);
    buf ^= 1;
  }

  const int hf = lane >> 4;
  const int nl = lane & 15;
#pragma unroll
  for (int i = 0; i < 2; ++i)
#pragma unroll
    for (int j = 0; j < 4; ++j)
#pragma unroll
      for (int v = 0; v < 8; ++v) {
        int rLoc = (mtBase + i) * 16 + v + hf * 8;
        int t = t0 + rLoc;
        int h = h0 + (ntBase + j) * 16 + nl;
        float cwv = cwLds[rLoc];
        size_t o = (size_t)t * H_DIM + h;
        float r = cwv * acc[i][j][v];
        out[o] = first ? r : (out[o] + r);   // expert 0 overwrites poison
      }
}

// ---------------------------------------------------------------------------
extern "C" void kernel_launch(void* const* d_in, const int* in_sizes, int n_in,
                              void* d_out, int out_size, void* d_ws, size_t ws_size,
                              hipStream_t stream)
{
  const float* x   = (const float*)d_in[0];   // [B,S,H]
  const float* wg  = (const float*)d_in[1];   // [H,E]
  const float* wgp = (const float*)d_in[2];   // [E,H,F]
  const float* wup = (const float*)d_in[3];   // [E,H,F]
  const float* wdp = (const float*)d_in[4];   // [E,F,H]

  float* out    = (float*)d_out;                        // [T,H]
  float* logits = out + (size_t)T_TOK * H_DIM;          // [T,E]

  // Workspace: cw [T,E] f32 (256 KB) + act [T,F] bf16 (64 MB, reused per expert)
  float* cw = (float*)d_ws;
  bf16* act = (bf16*)((char*)d_ws + (size_t)T_TOK * E_NUM * sizeof(float));

  router_kernel<<<T_TOK / 8, 256, 0, stream>>>(x, wg, cw, logits);

  for (int e = 0; e < E_NUM; ++e) {
    gate_up_kernel<<<dim3(T_TOK / BM, F_DIM / BN1), 256, 0, stream>>>(
        x, wgp + (size_t)e * H_DIM * F_DIM, wup + (size_t)e * H_DIM * F_DIM, act);
    down_kernel<<<dim3(T_TOK / BM, H_DIM / BN2), 256, 0, stream>>>(
        act, wdp + (size_t)e * F_DIM * H_DIM, cw, e, (e == 0) ? 1 : 0, out);
  }
}
